// Receiver_11029476016568
// MI455X (gfx1250) — matrix operand order repaired and checked by reading, not compiled, not run on hardware
//
#include <hip/hip_runtime.h>

// Problem constants (match reference)
#define B_ 1024
#define S_ 128
#define E_ 512
#define H_ 1024
#define N_ 4096

typedef __attribute__((ext_vector_type(16))) __bf16        v16bf;
typedef __attribute__((ext_vector_type(8)))  float         v8f;
typedef __attribute__((ext_vector_type(4)))  unsigned int  u32x4;
typedef __attribute__((ext_vector_type(8)))  int           i32x8;
typedef __attribute__((ext_vector_type(4)))  int           i32x4;
typedef __attribute__((ext_vector_type(2)))  unsigned int  u32x2;
typedef __attribute__((ext_vector_type(4)))  float         f32x4;

// TDM availability (this toolchain: 6-arg builtin).
#if defined(__has_builtin)
#if __has_builtin(__builtin_amdgcn_tensor_load_to_lds) && \
    __has_builtin(__builtin_amdgcn_s_wait_tensorcnt)
#define HAVE_TDM 1
#endif
#endif
#ifndef HAVE_TDM
#define HAVE_TDM 0
#endif

// fp32 -> bf16 round-to-nearest-even (bit trick)
static __device__ __forceinline__ unsigned short f2bf(float f) {
  union { float f; unsigned u; } x; x.f = f;
  unsigned r = x.u + 0x7FFFu + ((x.u >> 16) & 1u);
  return (unsigned short)(r >> 16);
}

union Pack32 { u32x4 q[2]; v16bf v; };

// A-matrix 16x32 bf16: base points at the lane's row; off is an
// element offset. One contiguous 32B chunk: [off, off+16).
static __device__ __forceinline__ v16bf load_a_off(const unsigned short* base, unsigned off) {
  Pack32 p;
  p.q[0] = *(const u32x4*)(base + (size_t)off);
  p.q[1] = *(const u32x4*)(base + (size_t)off + 8);
  return p.v;
}

// B-matrix 32x16 bf16 (B = W^T): scalar base + 32-bit per-lane element offset
// (GVS addressing). One contiguous 32B chunk of the weight row.
static __device__ __forceinline__ v16bf load_b_off(const unsigned short* base, unsigned off) {
  Pack32 p;
  const u32x4* q = (const u32x4*)(base + (size_t)off);
  p.q[0] = q[0];
  p.q[1] = q[1];
  return p.v;
}

#define WMMA_BF16(av, bv, cc) \
  __builtin_amdgcn_wmma_f32_16x16x32_bf16(false, (av), false, (bv), (short)0, (cc), false, false)

// ---------------------------------------------------------------------------
// Stage a contiguous run of bf16 elements from global into LDS via TDM.
// Wave 0 issues one 1D tensor_load_to_lds (TENSORcnt); D# per ISA 8.3/8.4:
// data_size=2B, tensor_dim0 = tile_dim0 = total_elems (<=65535), 1D.
static __device__ __forceinline__ void stage_issue(unsigned short* sA,
                                                   const unsigned short* gA,
                                                   int total_elems) {
#if HAVE_TDM
  if ((threadIdx.x >> 5) == 0) {
    unsigned lds_off = (unsigned)(unsigned long long)sA;   // LDS aperture: low 32 bits
    unsigned long long ga = (unsigned long long)gA;
    u32x4 g0 = {0u, 0u, 0u, 0u};
    g0[0] = 1u;                                            // count=1 (valid user D#)
    g0[1] = lds_off;                                       // lds_addr
    g0[2] = (unsigned)ga;                                  // global_addr[31:0]
    g0[3] = (unsigned)((ga >> 32) & 0x01FFFFFFull) | (2u << 30); // addr[56:32] | type=2
    i32x8 g1 = {0, 0, 0, 0, 0, 0, 0, 0};
    g1[0] = 1 << 16;                                       // data_size=1 -> 2 bytes
    g1[1] = (total_elems & 0xFFFF) << 16;                  // tensor_dim0[15:0]
    g1[2] = (total_elems >> 16) & 0xFFFF;                  // tensor_dim0[31:16]
    g1[3] = (total_elems & 0xFFFF) << 16;                  // tile_dim0
    g1[4] = 0;                                             // tile_dim1=0 (1D)
    g1[5] = total_elems;                                   // tensor_dim0_stride
    i32x4 z4 = {0, 0, 0, 0};
    i32x8 z8 = {0, 0, 0, 0, 0, 0, 0, 0};
    __builtin_amdgcn_tensor_load_to_lds(g0, g1, z4, z4, z8, 0);
  }
#else
  for (int i = threadIdx.x * 8; i < total_elems; i += blockDim.x * 8)
    *(u32x4*)(sA + i) = *(const u32x4*)(gA + i);
#endif
}

static __device__ __forceinline__ void stage_wait() {
#if HAVE_TDM
  if ((threadIdx.x >> 5) == 0) __builtin_amdgcn_s_wait_tensorcnt(0);
#endif
  __syncthreads();
}

// ---------------------------------------------------------------------------
__global__ void cvt_f32_bf16(const float* __restrict__ src,
                             unsigned short* __restrict__ dst, int npairs) {
  int i = blockIdx.x * blockDim.x + threadIdx.x;
  if (i >= npairs) return;
  float a = src[2 * i], b = src[2 * i + 1];
  ((unsigned*)dst)[i] = (unsigned)f2bf(a) | ((unsigned)f2bf(b) << 16);
}

__global__ void zero_u32(unsigned* __restrict__ p, int n) {
  int i = blockIdx.x * blockDim.x + threadIdx.x;
  if (i < n) p[i] = 0;
}

// Embedding gather: m[B,S] -> embs_bf16[S,B,E]; one block per (b,s).
__global__ void embed_bf16(const int* __restrict__ m,
                           const float* __restrict__ tbl,
                           unsigned short* __restrict__ out) {
  int p = blockIdx.x;
  int b = p >> 7;          // S_ == 128
  int s = p & 127;
  int tok = m[(size_t)b * S_ + s];
  const float* src = tbl + (size_t)tok * E_;
  unsigned short* dst = out + ((size_t)s * B_ + b) * E_;
  int t = threadIdx.x;
  f32x4 v = *(const f32x4*)(src + t * 4);
  u32x2 w;
  w.x = (unsigned)f2bf(v.x) | ((unsigned)f2bf(v.y) << 16);
  w.y = (unsigned)f2bf(v.z) | ((unsigned)f2bf(v.w) << 16);
  *(u32x2*)(dst + t * 4) = w;
}

// ---------------------------------------------------------------------------
// C[M,4096](f32) = A[M,K](bf16) @ W[4096,K]^T(bf16) + bias0 (+bias1)
// Block: 8 waves sharing one 16-row A tile (TDM-staged in LDS).
// Wave: 16x64 strip (4 N-tiles). Scalar W base (readfirstlane) + 32-bit
// per-lane offsets -> GVS loads; manual x2 ping-pong software pipeline.
template <int K>
__global__ void __launch_bounds__(256)
gemm_bf16_bias(const unsigned short* __restrict__ A,
               const unsigned short* __restrict__ W,
               const float* __restrict__ bias0,
               const float* __restrict__ bias1,
               float* __restrict__ C, int M) {
  __shared__ __attribute__((aligned(64))) unsigned short sA[16 * K];
  int wv = threadIdx.x >> 5;
  int lane = threadIdx.x & 31;
  int mt = blockIdx.x >> 3;                                  // 8 blocks / row tile
  int ng = __builtin_amdgcn_readfirstlane((blockIdx.x & 7) * 8 + wv); // wave-uniform
  int ln = lane & 15;
  int half = lane >> 4;
  (void)M;

  stage_issue(sA, A + (size_t)mt * 16 * K, 16 * K);

  v8f acc[4];
#pragma unroll
  for (int j = 0; j < 4; ++j) {
    int col = ng * 64 + j * 16 + ln;
    float bv = bias0 ? bias0[col] : 0.0f;
    if (bias1) bv += bias1[col];
#pragma unroll
    for (int r = 0; r < 8; ++r) acc[j][r] = bv;
  }

  const unsigned short* Wb = W + (size_t)ng * 64 * K;        // scalar (SGPR) base
  unsigned bo = (unsigned)(ln * K + half * 16);              // per-lane element off
  unsigned ao = (unsigned)(half * 16);                       // LDS element off

  v16bf a0, a1, b00, b01, b02, b03, b10, b11, b12, b13;
  b00 = load_b_off(Wb, bo + 0 * 16 * K);
  b01 = load_b_off(Wb, bo + 1 * 16 * K);
  b02 = load_b_off(Wb, bo + 2 * 16 * K);
  b03 = load_b_off(Wb, bo + 3 * 16 * K);

  stage_wait();
  const unsigned short* Arow = sA + ln * K;
  a0 = load_a_off(Arow, ao);

#pragma unroll 1
  for (int k = 0; k < K - 64; k += 64) {
    a1  = load_a_off(Arow, ao + k + 32);
    b10 = load_b_off(Wb, bo + k + 32 + 0 * 16 * K);
    b11 = load_b_off(Wb, bo + k + 32 + 1 * 16 * K);
    b12 = load_b_off(Wb, bo + k + 32 + 2 * 16 * K);
    b13 = load_b_off(Wb, bo + k + 32 + 3 * 16 * K);
    acc[0] = WMMA_BF16(a0, b00, acc[0]);
    acc[1] = WMMA_BF16(a0, b01, acc[1]);
    acc[2] = WMMA_BF16(a0, b02, acc[2]);
    acc[3] = WMMA_BF16(a0, b03, acc[3]);
    a0  = load_a_off(Arow, ao + k + 64);
    b00 = load_b_off(Wb, bo + k + 64 + 0 * 16 * K);
    b01 = load_b_off(Wb, bo + k + 64 + 1 * 16 * K);
    b02 = load_b_off(Wb, bo + k + 64 + 2 * 16 * K);
    b03 = load_b_off(Wb, bo + k + 64 + 3 * 16 * K);
    acc[0] = WMMA_BF16(a1, b10, acc[0]);
    acc[1] = WMMA_BF16(a1, b11, acc[1]);
    acc[2] = WMMA_BF16(a1, b12, acc[2]);
    acc[3] = WMMA_BF16(a1, b13, acc[3]);
  }
  // Epilogue: buffers0 hold chunk K-64; load and process final chunk K-32.
  a1  = load_a_off(Arow, ao + K - 32);
  b10 = load_b_off(Wb, bo + K - 32 + 0 * 16 * K);
  b11 = load_b_off(Wb, bo + K - 32 + 1 * 16 * K);
  b12 = load_b_off(Wb, bo + K - 32 + 2 * 16 * K);
  b13 = load_b_off(Wb, bo + K - 32 + 3 * 16 * K);
  acc[0] = WMMA_BF16(a0, b00, acc[0]);
  acc[1] = WMMA_BF16(a0, b01, acc[1]);
  acc[2] = WMMA_BF16(a0, b02, acc[2]);
  acc[3] = WMMA_BF16(a0, b03, acc[3]);
  acc[0] = WMMA_BF16(a1, b10, acc[0]);
  acc[1] = WMMA_BF16(a1, b11, acc[1]);
  acc[2] = WMMA_BF16(a1, b12, acc[2]);
  acc[3] = WMMA_BF16(a1, b13, acc[3]);

#pragma unroll
  for (int j = 0; j < 4; ++j) {
    int col = ng * 64 + j * 16 + ln;
#pragma unroll
    for (int r = 0; r < 8; ++r) {
      int row = mt * 16 + half * 8 + r;
      __builtin_nontemporal_store(acc[j][r], &C[(size_t)row * N_ + col]);  // 64-bit (C may exceed 2GB)
    }
  }
}

// ---------------------------------------------------------------------------
// One LSTM step: gates = x_gates[s] + h_in @ W_hh^T, activations, h/c update.
// Same pipelined structure; gate constants (g*H*H elems <= 6.3MB bytes) fold
// into the signed-24-bit instruction offsets. W_hh stays L2-resident.
__global__ void __launch_bounds__(256)
lstm_step(const float* __restrict__ xg,            // [B,4H] (this step's slice)
          const unsigned short* __restrict__ Whh,  // [4H,H]
          const unsigned short* __restrict__ h_in, // [B,H]
          unsigned short* __restrict__ h_out,      // [B,H]
          float* __restrict__ c) {                 // [B,H]
  __shared__ __attribute__((aligned(64))) unsigned short sA[16 * H_];
  int wv = threadIdx.x >> 5;
  int lane = threadIdx.x & 31;
  int mt = blockIdx.x >> 3;
  int nt = __builtin_amdgcn_readfirstlane((blockIdx.x & 7) * 8 + wv);
  int ln = lane & 15;
  int half = lane >> 4;

  stage_issue(sA, h_in + (size_t)mt * 16 * H_, 16 * H_);

  // Accumulators init from x_gates: 32-bit offsets within the 16MB step slice.
  v8f acc[4];
  unsigned xbase = (unsigned)((mt * 16 + half * 8) * (4 * H_) + nt * 16 + ln);
#pragma unroll
  for (int g = 0; g < 4; ++g) {
#pragma unroll
    for (int r = 0; r < 8; ++r)
      acc[g][r] = __builtin_nontemporal_load(xg + (size_t)(xbase + g * H_ + r * 4 * H_));
  }

  const unsigned short* Wb = Whh + (size_t)nt * 16 * H_;     // scalar base
  unsigned bo = (unsigned)(ln * H_ + half * 16);
  unsigned ao = (unsigned)(half * 16);

  v16bf a0, a1, b00, b01, b02, b03, b10, b11, b12, b13;
  b00 = load_b_off(Wb, bo + 0u * H_ * H_);
  b01 = load_b_off(Wb, bo + 1u * H_ * H_);
  b02 = load_b_off(Wb, bo + 2u * H_ * H_);
  b03 = load_b_off(Wb, bo + 3u * H_ * H_);

  stage_wait();
  const unsigned short* Arow = sA + ln * H_;
  a0 = load_a_off(Arow, ao);

#pragma unroll 1
  for (int k = 0; k < H_ - 64; k += 64) {
    a1  = load_a_off(Arow, ao + k + 32);
    b10 = load_b_off(Wb, bo + k + 32 + 0u * H_ * H_);
    b11 = load_b_off(Wb, bo + k + 32 + 1u * H_ * H_);
    b12 = load_b_off(Wb, bo + k + 32 + 2u * H_ * H_);
    b13 = load_b_off(Wb, bo + k + 32 + 3u * H_ * H_);
    acc[0] = WMMA_BF16(a0, b00, acc[0]);
    acc[1] = WMMA_BF16(a0, b01, acc[1]);
    acc[2] = WMMA_BF16(a0, b02, acc[2]);
    acc[3] = WMMA_BF16(a0, b03, acc[3]);
    a0  = load_a_off(Arow, ao + k + 64);
    b00 = load_b_off(Wb, bo + k + 64 + 0u * H_ * H_);
    b01 = load_b_off(Wb, bo + k + 64 + 1u * H_ * H_);
    b02 = load_b_off(Wb, bo + k + 64 + 2u * H_ * H_);
    b03 = load_b_off(Wb, bo + k + 64 + 3u * H_ * H_);
    acc[0] = WMMA_BF16(a1, b10, acc[0]);
    acc[1] = WMMA_BF16(a1, b11, acc[1]);
    acc[2] = WMMA_BF16(a1, b12, acc[2]);
    acc[3] = WMMA_BF16(a1, b13, acc[3]);
  }
  a1  = load_a_off(Arow, ao + H_ - 32);
  b10 = load_b_off(Wb, bo + H_ - 32 + 0u * H_ * H_);
  b11 = load_b_off(Wb, bo + H_ - 32 + 1u * H_ * H_);
  b12 = load_b_off(Wb, bo + H_ - 32 + 2u * H_ * H_);
  b13 = load_b_off(Wb, bo + H_ - 32 + 3u * H_ * H_);
  acc[0] = WMMA_BF16(a0, b00, acc[0]);
  acc[1] = WMMA_BF16(a0, b01, acc[1]);
  acc[2] = WMMA_BF16(a0, b02, acc[2]);
  acc[3] = WMMA_BF16(a0, b03, acc[3]);
  acc[0] = WMMA_BF16(a1, b10, acc[0]);
  acc[1] = WMMA_BF16(a1, b11, acc[1]);
  acc[2] = WMMA_BF16(a1, b12, acc[2]);
  acc[3] = WMMA_BF16(a1, b13, acc[3]);

  // C/D layout: lane = column (nt*16+ln), VGPR r = row (half*8+r).
#pragma unroll
  for (int r = 0; r < 8; ++r) {
    int row = mt * 16 + half * 8 + r;
    int col = nt * 16 + ln;
    size_t idx = (size_t)row * H_ + col;
    float iv = acc[0][r], fv = acc[1][r], gv = acc[2][r], ov = acc[3][r];
    iv = 1.0f / (1.0f + __expf(-iv));
    fv = 1.0f / (1.0f + __expf(-fv));
    gv = tanhf(gv);
    ov = 1.0f / (1.0f + __expf(-ov));
    float cn = fv * c[idx] + iv * gv;
    c[idx] = cn;
    h_out[idx] = f2bf(ov * tanhf(cn));
  }
}

// ---------------------------------------------------------------------------
extern "C" void kernel_launch(void* const* d_in, const int* in_sizes, int n_in,
                              void* d_out, int out_size, void* d_ws, size_t ws_size,
                              hipStream_t stream) {
  (void)in_sizes; (void)n_in; (void)out_size; (void)ws_size;
  const int*   m       = (const int*)d_in[0];
  const float* emb_tab = (const float*)d_in[1];
  const float* W_ih    = (const float*)d_in[2];
  const float* W_hh    = (const float*)d_in[3];
  const float* b_ih    = (const float*)d_in[4];
  const float* b_hh    = (const float*)d_in[5];
  const float* W_out   = (const float*)d_in[6];
  const float* b_out   = (const float*)d_in[7];
  float* out = (float*)d_out;

  char* ws = (char*)d_ws;
  size_t off = 0;
  auto take = [&](size_t bytes) {
    char* p = ws + off;
    off += (bytes + 255) & ~(size_t)255;
    return p;
  };
  unsigned short* Wih_bf  = (unsigned short*)take((size_t)2 * 4 * H_ * E_);
  unsigned short* Whh_bf  = (unsigned short*)take((size_t)2 * 4 * H_ * H_);
  unsigned short* Wout_bf = (unsigned short*)take((size_t)2 * N_ * H_);
  unsigned short* embs    = (unsigned short*)take((size_t)2 * S_ * B_ * E_);
  float*          xg      = (float*)take((size_t)4 * S_ * B_ * 4 * H_);
  unsigned short* h0      = (unsigned short*)take((size_t)2 * B_ * H_);
  unsigned short* h1      = (unsigned short*)take((size_t)2 * B_ * H_);
  float*          cbuf    = (float*)take((size_t)4 * B_ * H_);

  // 1) weights -> bf16
  cvt_f32_bf16<<<(4 * H_ * E_ / 2 + 255) / 256, 256, 0, stream>>>(W_ih, Wih_bf, 4 * H_ * E_ / 2);
  cvt_f32_bf16<<<(4 * H_ * H_ / 2 + 255) / 256, 256, 0, stream>>>(W_hh, Whh_bf, 4 * H_ * H_ / 2);
  cvt_f32_bf16<<<(N_ * H_ / 2 + 255) / 256, 256, 0, stream>>>(W_out, Wout_bf, N_ * H_ / 2);

  // 2) embedding gather -> bf16 time-major [S,B,E]
  embed_bf16<<<B_ * S_, 128, 0, stream>>>(m, emb_tab, embs);

  // 3) x_gates[S*B,4H] = embs @ W_ih^T + b_ih + b_hh   (K = E = 512)
  {
    int M = S_ * B_;
    int blocks = (M / 16) * 8;             // 65536 blocks, 8 waves each
    gemm_bf16_bias<E_><<<blocks, 256, 0, stream>>>(embs, Wih_bf, b_ih, b_hh, xg, M);
  }

  // 4) h0 = 0 (bf16), c = 0 (fp32)
  zero_u32<<<(B_ * H_ / 2 + 255) / 256, 256, 0, stream>>>((unsigned*)h0, B_ * H_ / 2);
  zero_u32<<<(B_ * H_ + 255) / 256, 256, 0, stream>>>((unsigned*)cbuf, B_ * H_);

  // 5) recurrence: 128 steps, double-buffered h
  unsigned short* hin = h0;
  unsigned short* hout = h1;
  for (int s = 0; s < S_; ++s) {
    int blocks = (B_ / 16) * 8;            // 512 blocks
    lstm_step<<<blocks, 256, 0, stream>>>(xg + (size_t)s * B_ * 4 * H_,
                                          Whh_bf, hin, hout, cbuf);
    unsigned short* t = hin; hin = hout; hout = t;
  }

  // 6) logits = h_final @ W_out^T + b_out   (K = H = 1024)
  {
    int M = B_;
    int blocks = (M / 16) * 8;             // 512 blocks
    gemm_bf16_bias<H_><<<blocks, 256, 0, stream>>>(hin, Wout_bf, b_out, nullptr, out, M);
  }
}
